// Flux_41274635714824
// MI455X (gfx1250) — compile-verified, matrix operand order checked by reading, not run on hardware
//
#include <hip/hip_runtime.h>
#include <hip/hip_bf16.h>

// ---------------------------------------------------------------------------
// Types / helpers
// ---------------------------------------------------------------------------
typedef float  v8f   __attribute__((ext_vector_type(8)));
typedef __bf16 v16bf __attribute__((ext_vector_type(16)));
typedef __bf16 v2bf  __attribute__((ext_vector_type(2)));

union AccU  { v8f v;   float f[8]; };
union FragU { v16bf v; uint4 q[2]; unsigned w[8]; unsigned short u[16]; };

__device__ inline unsigned short f2bfbits(float f) {
  unsigned int u = __float_as_uint(f);
  unsigned int r = u + 0x7FFFu + ((u >> 16) & 1u);   // round-to-nearest-even
  return (unsigned short)(r >> 16);
}

// packed f32x2 -> bf16x2 (v_cvt_pk_bf16_f32 when available)
__device__ inline unsigned pk_bf16(float x, float y) {
#if __has_builtin(__builtin_amdgcn_cvt_pk_bf16_f32)
  v2bf r = __builtin_amdgcn_cvt_pk_bf16_f32(x, y);
  return __builtin_bit_cast(unsigned, r);
#else
  return (unsigned)f2bfbits(x) | ((unsigned)f2bfbits(y) << 16);
#endif
}

__device__ inline v16bf ldfrag(const unsigned short* p0, const unsigned short* p1) {
  FragU u;
  u.q[0] = *(const uint4*)p0;
  u.q[1] = *(const uint4*)p1;
  return u.v;
}

__device__ inline v8f wmma_bf16(v16bf a, v16bf b, v8f c) {
  return __builtin_amdgcn_wmma_f32_16x16x32_bf16(false, a, false, b, (short)0, c, false, false);
}

// async copy 16B global -> LDS (per lane), tracked by ASYNCcnt
__device__ inline void async_g2l_b128(unsigned lds_addr, const void* gaddr) {
  asm volatile("global_load_async_to_lds_b128 %0, %1, off"
               :: "v"(lds_addr), "v"(gaddr) : "memory");
}
__device__ inline void wait_asynccnt0() {
  asm volatile("s_wait_asynccnt 0x0" ::: "memory");
}
__device__ inline void wait_asynccnt2() {
  asm volatile("s_wait_asynccnt 0x2" ::: "memory");
}
__device__ inline unsigned lds_addr_of(const void* p) {
  // generic pointer to LDS: addr[31:0] is the wave-relative LDS byte offset
  return (unsigned)(unsigned long long)p;
}

__device__ inline float geluf(float x) {
  return 0.5f * x * (1.f + tanhf(0.7978845608028654f * (x + 0.044715f * x * x * x)));
}
__device__ inline float siluf(float x) { return x / (1.f + expf(-x)); }

// ---------------------------------------------------------------------------
// WMMA GEMM: C[M,N] = act(A[M,K] * W[K,N] + bias[N])
// A rows may be batch-split: row r -> batch b=r/rowsPerB, A + b*strideA + (r%rowsPerB)*lda
// Block: 256 threads (8 waves), tile 128x64, BK=32.
// Double-buffered LDS (bf16); packed v_cvt_pk_bf16_f32 staging; next-tile
// global loads overlap current-tile WMMAs; W prefetch two tiles ahead.
// ---------------------------------------------------------------------------
__global__ __launch_bounds__(256) void k_gemm(
    const float* __restrict__ A, long long strideA, int lda, int rowsPerB,
    const float* __restrict__ W, int ldw, const float* __restrict__ bias,
    float* __restrict__ C, long long strideC, int ldc,
    int M, int N, int K, int act)
{
  __shared__ __align__(16) unsigned short As[2][128 * 32];   // [buf][row][k]
  __shared__ __align__(16) unsigned short BsT[2][64 * 32];   // [buf][col][k]
  const int tid  = threadIdx.x;
  const int m0   = blockIdx.y * 128;
  const int n0   = blockIdx.x * 64;
  const int lane = tid & 31, lh = lane & 15, ks = (lane >> 4) << 3;
  const int wid  = tid >> 5;
  const int wm   = (wid & 3) << 5;   // 4 waves in M
  const int wn   = (wid >> 2) << 5;  // 2 waves in N

  AccU acc[2][2];
#pragma unroll
  for (int mi = 0; mi < 2; ++mi)
#pragma unroll
    for (int ni = 0; ni < 2; ++ni)
#pragma unroll
      for (int j = 0; j < 8; ++j) acc[mi][ni].f[j] = 0.f;

  float4 ra[4];        // staged A items (row = f>>3, kq = (f&7)*4)
  float2 rb[2][2];     // staged B 2x2 blocks (kp = f>>5, np = f&31)

  auto load_tiles = [&](int k0) {
#pragma unroll
    for (int i = 0; i < 4; ++i) {
      int f = tid + (i << 8);
      int row = f >> 3, kq = (f & 7) << 2;
      int gr = m0 + row;
      ra[i] = make_float4(0.f, 0.f, 0.f, 0.f);
      if (gr < M) {
        int bb = gr / rowsPerB, lr = gr - bb * rowsPerB;
        ra[i] = *(const float4*)(A + (size_t)bb * strideA + (size_t)lr * lda + k0 + kq);
      }
    }
#pragma unroll
    for (int i = 0; i < 2; ++i) {
      int f = tid + (i << 8);
      int kp = f >> 5, np = f & 31;
      const float* wp = W + (size_t)(k0 + 2 * kp) * ldw + n0 + 2 * np;
      rb[i][0] = *(const float2*)wp;
      rb[i][1] = *(const float2*)(wp + ldw);
    }
  };
  auto store_tiles = [&](int bsel) {
#pragma unroll
    for (int i = 0; i < 4; ++i) {
      int f = tid + (i << 8);
      int row = f >> 3, kq = (f & 7) << 2;
      *(uint2*)&As[bsel][row * 32 + kq] =
          make_uint2(pk_bf16(ra[i].x, ra[i].y), pk_bf16(ra[i].z, ra[i].w));
    }
#pragma unroll
    for (int i = 0; i < 2; ++i) {
      int f = tid + (i << 8);
      int kp = f >> 5, np = f & 31;
      *(unsigned*)&BsT[bsel][(2 * np) * 32 + 2 * kp]     = pk_bf16(rb[i][0].x, rb[i][1].x);
      *(unsigned*)&BsT[bsel][(2 * np + 1) * 32 + 2 * kp] = pk_bf16(rb[i][0].y, rb[i][1].y);
    }
  };

  const int nsteps = K >> 5;
  load_tiles(0);
  for (int s = 0; s < nsteps; ++s) {
    store_tiles(s & 1);
    __syncthreads();                       // buf[s&1] visible to all waves
    if (s + 1 < nsteps) {
      load_tiles((s + 1) << 5);            // overlap with this step's WMMAs
      if (s + 2 < nsteps)
        __builtin_prefetch(W + (size_t)(((s + 2) << 5) + (tid >> 3)) * ldw +
                               n0 + ((tid & 7) << 3), 0, 1);
    }
    const unsigned short* Ab = As[s & 1];
    const unsigned short* Bb = BsT[s & 1];
    v16bf af[2], bfv[2];
#pragma unroll
    for (int mi = 0; mi < 2; ++mi) {
      int r = wm + mi * 16 + lh;
      af[mi] = ldfrag(&Ab[r * 32 + ks], &Ab[r * 32 + ks + 16]);
    }
#pragma unroll
    for (int ni = 0; ni < 2; ++ni) {
      int cN = wn + ni * 16 + lh;
      bfv[ni] = ldfrag(&Bb[cN * 32 + ks], &Bb[cN * 32 + ks + 16]);
    }
#pragma unroll
    for (int mi = 0; mi < 2; ++mi)
#pragma unroll
      for (int ni = 0; ni < 2; ++ni)
        acc[mi][ni].v = wmma_bf16(af[mi], bfv[ni], acc[mi][ni].v);
    if (s + 1 < nsteps) __syncthreads();   // done reading buf[(s+1)&1] (from s-1)
  }

  const int hi8 = (lane >> 4) << 3;
#pragma unroll
  for (int mi = 0; mi < 2; ++mi) {
#pragma unroll
    for (int ni = 0; ni < 2; ++ni) {
      int colg = n0 + wn + ni * 16 + lh;
      if (colg >= N) continue;
      float bv = bias ? bias[colg] : 0.f;
#pragma unroll
      for (int j = 0; j < 8; ++j) {
        int rowg = m0 + wm + mi * 16 + j + hi8;
        if (rowg < M) {
          float val = acc[mi][ni].f[j] + bv;
          if (act == 1) val = geluf(val);
          int bb = rowg / rowsPerB, lr = rowg - bb * rowsPerB;
          C[(size_t)bb * strideC + (size_t)lr * ldc + colg] = val;
        }
      }
    }
  }
}

// ---------------------------------------------------------------------------
// Flash attention: 8 waves per block, one (b,h), 128 queries per block.
// K/V tiles (bf16, 32 keys) double-buffered in LDS via
// global_load_async_to_lds_b128: step i+1's DMA runs under step i's WMMAs;
// s_wait_asynccnt 2 (in-order async completion) gates the current tile.
// q,k: bf16 [B,H,L,64] (q pre-scaled by 1/8); v: bf16 transposed [B,H,64,L].
// Computes S^T = K*Q^T so the C-tile per-lane data is exactly the B-fragment
// layout for P^T in O^T = V^T * P^T. out: f32 [B,L,ldo] at column h*64.
// ---------------------------------------------------------------------------
__global__ __launch_bounds__(256) void k_attn(
    const unsigned short* __restrict__ qb, const unsigned short* __restrict__ kb,
    const unsigned short* __restrict__ vT, float* __restrict__ out,
    int ldo, int L, int H)
{
  __shared__ __align__(16) unsigned short Ks[2][32 * 64];  // [buf][key][d]
  __shared__ __align__(16) unsigned short Vs[2][64 * 32];  // [buf][d][key]
  const int tid = threadIdx.x;
  const int wid = tid >> 5, lane = tid & 31;
  const int lh = lane & 15, ks = (lane >> 4) << 3;
  const int h = blockIdx.y, b = blockIdx.z;
  const int qt = blockIdx.x * 8 + wid;
  const size_t bh = (size_t)b * H + h;

  const unsigned short* qrow = qb + (bh * L + (size_t)qt * 16 + lh) * 64;
  v16bf qf0 = ldfrag(qrow + ks, qrow + ks + 16);
  v16bf qf1 = ldfrag(qrow + 32 + ks, qrow + 32 + ks + 16);

  // async staging assignment: each wave copies 4 K-rows (4x128B) + 8 V-rows (8x64B)
  const int krow_l = 4 * wid + (lane >> 3);
  const int kchunk = (lane & 7) << 3;
  const int vrow_l = 8 * wid + (lane >> 2);
  const int vchunk = (lane & 3) << 3;

  const unsigned ks_lds[2] = { lds_addr_of(&Ks[0][krow_l * 64 + kchunk]),
                               lds_addr_of(&Ks[1][krow_l * 64 + kchunk]) };
  const unsigned vs_lds[2] = { lds_addr_of(&Vs[0][vrow_l * 32 + vchunk]),
                               lds_addr_of(&Vs[1][vrow_l * 32 + vchunk]) };
  const unsigned short* kg_base = kb + (bh * L + krow_l) * 64 + kchunk;
  const unsigned short* vg_base = vT + (bh * 64 + vrow_l) * L + vchunk;

  AccU acc[4];
#pragma unroll
  for (int c = 0; c < 4; ++c)
#pragma unroll
    for (int j = 0; j < 8; ++j) acc[c].f[j] = 0.f;

  float mprev = -1e30f, lsum = 0.f;

  const int nst = L >> 5;
  async_g2l_b128(ks_lds[0], kg_base);
  async_g2l_b128(vs_lds[0], vg_base);

  for (int i = 0; i < nst; ++i) {
    const int kt = i << 5;
    if (i + 1 < nst) {
      async_g2l_b128(ks_lds[(i + 1) & 1], kg_base + ((size_t)kt + 32) * 64);
      async_g2l_b128(vs_lds[(i + 1) & 1], vg_base + kt + 32);
      wait_asynccnt2();   // oldest 2 (this step's tile) complete; next tile in flight
    } else {
      wait_asynccnt0();
    }
    __syncthreads();      // tile i resident for all waves

    const unsigned short* Kb = Ks[i & 1];
    const unsigned short* Vb = Vs[i & 1];

    AccU st[2];
#pragma unroll
    for (int s = 0; s < 2; ++s) {
      const unsigned short* krow = &Kb[(s * 16 + lh) * 64];
      v16bf ka0 = ldfrag(krow + ks, krow + ks + 16);
      v16bf ka1 = ldfrag(krow + 32 + ks, krow + 32 + ks + 16);
      v8f z = {0.f, 0.f, 0.f, 0.f, 0.f, 0.f, 0.f, 0.f};
      z = wmma_bf16(ka0, qf0, z);
      z = wmma_bf16(ka1, qf1, z);
      st[s].v = z;
    }
    // per-query (per-lane + partner lane) online softmax
    float rm = -1e30f;
#pragma unroll
    for (int s = 0; s < 2; ++s)
#pragma unroll
      for (int j = 0; j < 8; ++j) rm = fmaxf(rm, st[s].f[j]);
    rm = fmaxf(rm, __shfl_xor(rm, 16, 32));
    float mnew = fmaxf(mprev, rm);
    float corr = expf(mprev - mnew);
    float rs = 0.f;
    FragU pb;
#pragma unroll
    for (int s = 0; s < 2; ++s)
#pragma unroll
      for (int jj = 0; jj < 4; ++jj) {
        float p0 = expf(st[s].f[2 * jj]     - mnew);
        float p1 = expf(st[s].f[2 * jj + 1] - mnew);
        rs += p0 + p1;
        pb.w[s * 4 + jj] = pk_bf16(p0, p1);
      }
    rs += __shfl_xor(rs, 16, 32);
    lsum = lsum * corr + rs;
    mprev = mnew;

#pragma unroll
    for (int c = 0; c < 4; ++c) {
#pragma unroll
      for (int j = 0; j < 8; ++j) acc[c].f[j] *= corr;
      const unsigned short* vrow = &Vb[(c * 16 + lh) * 32];
      v16bf vf = ldfrag(vrow + ks, vrow + ks + 16);
      acc[c].v = wmma_bf16(vf, pb.v, acc[c].v);
    }
    __syncthreads();      // all waves done reading buf i&1 before it is re-filled
  }

  float inv = 1.f / lsum;
  float* orow = out + ((size_t)b * L + (size_t)qt * 16 + lh) * ldo + h * 64;
#pragma unroll
  for (int c = 0; c < 4; ++c)
#pragma unroll
    for (int j = 0; j < 8; ++j)
      orow[c * 16 + j + ks] = acc[c].f[j] * inv;
}

// ---------------------------------------------------------------------------
// QKV prep: split heads, RMS-norm q/k, apply RoPE, fold 1/sqrt(64) into q,
// write q,k bf16 [B,H,Ltot,64] (packed pair stores); v transposed [B,H,64,Ltot].
// ---------------------------------------------------------------------------
__global__ void k_qkvprep(const float* __restrict__ src, int ld, int Lsrc, int loff,
                          const float* __restrict__ qsc, const float* __restrict__ ksc,
                          const float* __restrict__ cosb, const float* __restrict__ sinb,
                          unsigned short* __restrict__ qb, unsigned short* __restrict__ kb,
                          unsigned short* __restrict__ vT, int Ltot, int H)
{
  int t = blockIdx.x * blockDim.x + threadIdx.x;
  int total = 2 * H * Lsrc;
  if (t >= total) return;
  int l = t % Lsrc;
  int h = (t / Lsrc) % H;
  int b = t / (Lsrc * H);
  const float* row = src + ((size_t)b * Lsrc + l) * ld;
  const float* qp = row + h * 64;
  const float* kp = row + 1024 + h * 64;
  const float* vp = row + 2048 + h * 64;
  int lg = loff + l;
  const float* cr = cosb + ((size_t)b * Ltot + lg) * 32;
  const float* sr = sinb + ((size_t)b * Ltot + lg) * 32;

  float qv[64], kv[64];
  float sq = 0.f, sk = 0.f;
#pragma unroll
  for (int d = 0; d < 64; ++d) {
    qv[d] = qp[d]; sq += qv[d] * qv[d];
    kv[d] = kp[d]; sk += kv[d] * kv[d];
  }
  float rq = rsqrtf(sq * (1.f / 64.f) + 1e-6f);
  float rk = rsqrtf(sk * (1.f / 64.f) + 1e-6f);
#pragma unroll
  for (int d = 0; d < 64; ++d) {
    qv[d] *= rq * qsc[d];
    kv[d] *= rk * ksc[d];
  }
  size_t obase = ((size_t)(b * H + h) * Ltot + lg) * 64;
#pragma unroll
  for (int j = 0; j < 32; ++j) {
    float c = cr[j], s = sr[j];
    float q0 = qv[2 * j], q1 = qv[2 * j + 1];
    float k0 = kv[2 * j], k1 = kv[2 * j + 1];
    *(unsigned*)&qb[obase + 2 * j] =
        pk_bf16((c * q0 - s * q1) * 0.125f, (s * q0 + c * q1) * 0.125f);
    *(unsigned*)&kb[obase + 2 * j] =
        pk_bf16(c * k0 - s * k1, s * k0 + c * k1);
  }
  size_t vbase = ((size_t)(b * H + h) * 64) * Ltot + lg;
#pragma unroll
  for (int d = 0; d < 64; ++d)
    vT[vbase + (size_t)d * Ltot] = f2bfbits(vp[d]);
}

// ---------------------------------------------------------------------------
// LayerNorm (C=1024) with modulation: y = (1+sc)*(LN(x)*g + be) + sh
// ---------------------------------------------------------------------------
__global__ __launch_bounds__(256) void k_ln_mod(
    const float* __restrict__ x, int bstride, int Lrows, float* __restrict__ y,
    const float* __restrict__ g, const float* __restrict__ be,
    const float* __restrict__ mod, int mstride, int shoff, int scoff)
{
  int row = blockIdx.x;
  int b = row / Lrows, l = row - b * Lrows;
  const float* xr = x + ((size_t)b * bstride + l) * 1024;
  float* yr = y + (size_t)row * 1024;
  int tid = threadIdx.x;
  __shared__ float red[256];
  __shared__ float smean, svar;
  float v[4];
  float s = 0.f;
#pragma unroll
  for (int i = 0; i < 4; ++i) { v[i] = xr[tid + i * 256]; s += v[i]; }
  red[tid] = s; __syncthreads();
  for (int o = 128; o > 0; o >>= 1) { if (tid < o) red[tid] += red[tid + o]; __syncthreads(); }
  if (tid == 0) smean = red[0] * (1.f / 1024.f);
  __syncthreads();
  float mean = smean;
  float s2 = 0.f;
#pragma unroll
  for (int i = 0; i < 4; ++i) { float d = v[i] - mean; s2 += d * d; }
  red[tid] = s2; __syncthreads();
  for (int o = 128; o > 0; o >>= 1) { if (tid < o) red[tid] += red[tid + o]; __syncthreads(); }
  if (tid == 0) svar = red[0] * (1.f / 1024.f);
  __syncthreads();
  float inv = rsqrtf(svar + 1e-6f);
  const float* shv = mod + (size_t)b * mstride + shoff;
  const float* scv = mod + (size_t)b * mstride + scoff;
#pragma unroll
  for (int i = 0; i < 4; ++i) {
    int c = tid + i * 256;
    float n = (v[i] - mean) * inv * g[c] + be[c];
    yr[c] = (1.f + scv[c]) * n + shv[c];
  }
}

// ---------------------------------------------------------------------------
// Small-M dense (vec paths, modulations): C[M,N] = act(opt_silu(A) @ W + b)
// ---------------------------------------------------------------------------
__global__ void k_dense_small(const float* __restrict__ A, int insilu,
                              const float* __restrict__ W, const float* __restrict__ bias,
                              float* __restrict__ C, int accum, int actout,
                              int M, int N, int K)
{
  int idx = blockIdx.x * blockDim.x + threadIdx.x;
  if (idx >= M * N) return;
  int m = idx / N, n = idx - m * N;
  float s = bias[n];
  for (int k = 0; k < K; ++k) {
    float a = A[(size_t)m * K + k];
    if (insilu) a = siluf(a);
    s += a * W[(size_t)k * N + n];
  }
  if (actout == 1) s = siluf(s);
  if (accum) C[idx] += s; else C[idx] = s;
}

__global__ void k_timeemb(const float* __restrict__ t, float* __restrict__ out)
{
  int idx = blockIdx.x * blockDim.x + threadIdx.x;
  if (idx >= 512) return;
  int b = idx >> 8, j = idx & 255;
  float tv = t[b] * 1000.f;
  int half = j & 127;
  float f = expf(-logf(10000.f) * (float)half / 128.f);
  float a = tv * f;
  out[idx] = (j < 128) ? cosf(a) : sinf(a);
}

__global__ void k_rope_tab(const int* __restrict__ img_ids, const int* __restrict__ txt_ids,
                           float* __restrict__ cosb, float* __restrict__ sinb)
{
  int idx = blockIdx.x * blockDim.x + threadIdx.x;
  if (idx >= 2 * 1280 * 32) return;
  int j = idx & 31;
  int l = (idx >> 5) % 1280;
  int b = idx / (1280 * 32);
  int axis, jp, dim;
  if (j < 8)       { axis = 0; jp = j;      dim = 16; }
  else if (j < 20) { axis = 1; jp = j - 8;  dim = 24; }
  else             { axis = 2; jp = j - 20; dim = 24; }
  float idv;
  if (l < 256) idv = (float)txt_ids[((size_t)b * 256 + l) * 3 + axis];
  else         idv = (float)img_ids[((size_t)b * 1024 + (l - 256)) * 3 + axis];
  float omega = expf(-logf(10000.f) * (2.f * jp) / (float)dim);
  float a = idv * omega;
  cosb[idx] = cosf(a);
  sinb[idx] = sinf(a);
}

__global__ void k_resid(float* __restrict__ x, const float* __restrict__ y,
                        const float* __restrict__ mod, int mstride, int goff,
                        int rowsPerB, long long total)
{
  long long idx = (long long)blockIdx.x * blockDim.x + threadIdx.x;
  if (idx >= total) return;
  int row = (int)(idx >> 10);
  int c = (int)(idx & 1023);
  int b = row / rowsPerB;
  x[idx] += mod[(size_t)b * mstride + goff + c] * y[idx];
}

__global__ void k_build_x(const float* __restrict__ txt, const float* __restrict__ img,
                          float* __restrict__ x)
{
  long long idx = (long long)blockIdx.x * blockDim.x + threadIdx.x;
  if (idx >= (long long)2 * 1280 * 1024) return;
  int c = (int)(idx & 1023);
  int r = (int)((idx >> 10) % 1280);
  int b = (int)(idx / (1280 * 1024));
  float v;
  if (r < 256) v = txt[((size_t)b * 256 + r) * 1024 + c];
  else         v = img[((size_t)b * 1024 + (r - 256)) * 1024 + c];
  x[idx] = v;
}

// ---------------------------------------------------------------------------
// Host orchestration
// ---------------------------------------------------------------------------
extern "C" void kernel_launch(void* const* d_in, const int* in_sizes, int n_in,
                              void* d_out, int out_size, void* d_ws, size_t ws_size,
                              hipStream_t stream)
{
  auto F = [&](int i) { return (const float*)d_in[i]; };
  const int* img_ids = (const int*)d_in[1];
  const int* txt_ids = (const int*)d_in[3];

  // workspace carve (deterministic)
  size_t off = 0;
  auto carve = [&](size_t bytes) -> char* {
    off = (off + 255) & ~(size_t)255;
    char* p = (char*)d_ws + off; off += bytes; return p;
  };
  float* vecb  = (float*)carve((size_t)2 * 1024 * 4);
  float* t256  = (float*)carve((size_t)2 * 256 * 4);
  float* tmph  = (float*)carve((size_t)2 * 1024 * 4);
  float* cosb  = (float*)carve((size_t)2 * 1280 * 32 * 4);
  float* sinb  = (float*)carve((size_t)2 * 1280 * 32 * 4);
  float* imgx  = (float*)carve((size_t)2 * 1024 * 1024 * 4);
  float* txtx  = (float*)carve((size_t)2 * 256 * 1024 * 4);
  float* xbuf  = (float*)carve((size_t)2 * 1280 * 1024 * 4);
  float* mbuf  = (float*)carve((size_t)2 * 1280 * 1024 * 4);
  float* modi  = (float*)carve((size_t)2 * 6144 * 4);
  float* modt  = (float*)carve((size_t)2 * 6144 * 4);
  float* mods  = (float*)carve((size_t)2 * 3072 * 4);
  float* adab  = (float*)carve((size_t)2 * 2048 * 4);
  float* h1    = (float*)carve((size_t)2 * 1280 * 7168 * 4);
  float* mlpb  = (float*)carve((size_t)2560 * 4096 * 4);
  float* attnb = (float*)carve((size_t)2 * 1280 * 1024 * 4);
  float* catb  = (float*)carve((size_t)2 * 1280 * 5120 * 4);
  float* tmpo  = (float*)carve((size_t)2 * 1280 * 1024 * 4);
  unsigned short* qbb = (unsigned short*)carve((size_t)2 * 16 * 1280 * 64 * 2);
  unsigned short* kbb = (unsigned short*)carve((size_t)2 * 16 * 1280 * 64 * 2);
  unsigned short* vtb = (unsigned short*)carve((size_t)2 * 16 * 1280 * 64 * 2);
  float* qkv_img = h1;
  float* qkv_txt = h1 + (size_t)2048 * 3072;

  auto gemm = [&](const float* A, long long sA, int lda, int rpb,
                  const float* W, int ldw, const float* bias,
                  float* C, long long sC, int ldc, int M, int N, int K, int act) {
    dim3 g((N + 63) / 64, (M + 127) / 128);
    k_gemm<<<g, dim3(256), 0, stream>>>(A, sA, lda, rpb, W, ldw, bias, C, sC, ldc, M, N, K, act);
  };
  auto dsmall = [&](const float* A, int insilu, const float* W, const float* bias,
                    float* C, int acc, int act, int M, int N, int K) {
    int n = M * N;
    k_dense_small<<<(n + 255) / 256, 256, 0, stream>>>(A, insilu, W, bias, C, acc, act, M, N, K);
  };
  auto ln = [&](const float* x, int bstr, int Lr, float* y, const float* g, const float* be,
                const float* mod, int mstr, int sho, int sco) {
    k_ln_mod<<<dim3(2 * Lr), dim3(256), 0, stream>>>(x, bstr, Lr, y, g, be, mod, mstr, sho, sco);
  };
  auto resid = [&](float* x, const float* y, const float* mod, int mstr, int goff,
                   int rpb, int rows) {
    long long n = (long long)rows * 1024;
    k_resid<<<(unsigned)((n + 255) / 256), 256, 0, stream>>>(x, y, mod, mstr, goff, rpb, n);
  };

  // ---- vec = mlp(time) + mlp(guidance) + mlp(y) ----
  k_timeemb<<<2, 256, 0, stream>>>(F(4), t256);
  dsmall(t256, 0, F(11), F(12), tmph, 0, 1, 2, 1024, 256);
  dsmall(tmph, 0, F(13), F(14), vecb, 0, 0, 2, 1024, 1024);
  k_timeemb<<<2, 256, 0, stream>>>(F(6), t256);
  dsmall(t256, 0, F(15), F(16), tmph, 0, 1, 2, 1024, 256);
  dsmall(tmph, 0, F(17), F(18), vecb, 1, 0, 2, 1024, 1024);
  dsmall(F(5), 0, F(19), F(20), tmph, 0, 1, 2, 1024, 768);
  dsmall(tmph, 0, F(21), F(22), vecb, 1, 0, 2, 1024, 1024);

  // ---- RoPE tables ----
  { int n = 2 * 1280 * 32; k_rope_tab<<<(n + 255) / 256, 256, 0, stream>>>(img_ids, txt_ids, cosb, sinb); }

  // ---- input projections ----
  gemm(F(0), 0, 64, 2048, F(7), 1024, F(8), imgx, 0, 1024, 2048, 1024, 64, 0);
  gemm(F(2), 0, 4096, 512, F(9), 1024, F(10), txtx, 0, 1024, 512, 1024, 4096, 0);

  // ---- double blocks ----
  for (int d = 0; d < 2; ++d) {
    int pi = 23 + d * 32;   // img params
    int pt = pi + 16;       // txt params
    dsmall(vecb, 1, F(pi + 0), F(pi + 1), modi, 0, 0, 2, 6144, 1024);
    dsmall(vecb, 1, F(pt + 0), F(pt + 1), modt, 0, 0, 2, 6144, 1024);

    ln(imgx, 1024, 1024, mbuf, F(pi + 8), F(pi + 9), modi, 6144, 0, 1024);
    gemm(mbuf, 0, 1024, 2048, F(pi + 2), 3072, F(pi + 3), qkv_img, 0, 3072, 2048, 3072, 1024, 0);
    ln(txtx, 256, 256, mbuf, F(pt + 8), F(pt + 9), modt, 6144, 0, 1024);
    gemm(mbuf, 0, 1024, 512, F(pt + 2), 3072, F(pt + 3), qkv_txt, 0, 3072, 512, 3072, 1024, 0);

    { int n = 2 * 16 * 256;
      k_qkvprep<<<(n + 255) / 256, 256, 0, stream>>>(qkv_txt, 3072, 256, 0, F(pt + 6), F(pt + 7),
                                                     cosb, sinb, qbb, kbb, vtb, 1280, 16); }
    { int n = 2 * 16 * 1024;
      k_qkvprep<<<(n + 255) / 256, 256, 0, stream>>>(qkv_img, 3072, 1024, 256, F(pi + 6), F(pi + 7),
                                                     cosb, sinb, qbb, kbb, vtb, 1280, 16); }
    k_attn<<<dim3(10, 16, 2), 256, 0, stream>>>(qbb, kbb, vtb, attnb, 1024, 1280, 16);

    gemm(attnb + (size_t)256 * 1024, (long long)1280 * 1024, 1024, 1024,
         F(pi + 4), 1024, F(pi + 5), tmpo, 0, 1024, 2048, 1024, 1024, 0);
    resid(imgx, tmpo, modi, 6144, 2 * 1024, 1024, 2048);
    gemm(attnb, (long long)1280 * 1024, 1024, 256,
         F(pt + 4), 1024, F(pt + 5), tmpo, 0, 1024, 512, 1024, 1024, 0);
    resid(txtx, tmpo, modt, 6144, 2 * 1024, 256, 512);

    ln(imgx, 1024, 1024, mbuf, F(pi + 10), F(pi + 11), modi, 6144, 3 * 1024, 4 * 1024);
    gemm(mbuf, 0, 1024, 2048, F(pi + 12), 4096, F(pi + 13), mlpb, 0, 4096, 2048, 4096, 1024, 1);
    gemm(mlpb, 0, 4096, 2048, F(pi + 14), 1024, F(pi + 15), tmpo, 0, 1024, 2048, 1024, 4096, 0);
    resid(imgx, tmpo, modi, 6144, 5 * 1024, 1024, 2048);

    ln(txtx, 256, 256, mbuf, F(pt + 10), F(pt + 11), modt, 6144, 3 * 1024, 4 * 1024);
    gemm(mbuf, 0, 1024, 512, F(pt + 12), 4096, F(pt + 13), mlpb, 0, 4096, 512, 4096, 1024, 1);
    gemm(mlpb, 0, 4096, 512, F(pt + 14), 1024, F(pt + 15), tmpo, 0, 1024, 512, 1024, 4096, 0);
    resid(txtx, tmpo, modt, 6144, 5 * 1024, 256, 512);
  }

  // ---- x = concat(txt, img) ----
  { long long n = (long long)2 * 1280 * 1024;
    k_build_x<<<(unsigned)((n + 255) / 256), 256, 0, stream>>>(txtx, imgx, xbuf); }

  // ---- single blocks ----
  for (int s = 0; s < 2; ++s) {
    int ps = 87 + s * 10;
    dsmall(vecb, 1, F(ps + 0), F(ps + 1), mods, 0, 0, 2, 3072, 1024);
    ln(xbuf, 1280, 1280, mbuf, F(ps + 8), F(ps + 9), mods, 3072, 0, 1024);
    // lin1 qkv half
    gemm(mbuf, 0, 1024, 2560, F(ps + 2), 7168, F(ps + 3), h1, 0, 3072, 2560, 3072, 1024, 0);
    // lin1 mlp half, gelu fused, written straight into lin2 input at col 1024
    gemm(mbuf, 0, 1024, 2560, F(ps + 2) + 3072, 7168, F(ps + 3) + 3072,
         catb + 1024, 0, 5120, 2560, 4096, 1024, 1);
    { int n = 2 * 16 * 1280;
      k_qkvprep<<<(n + 255) / 256, 256, 0, stream>>>(h1, 3072, 1280, 0, F(ps + 6), F(ps + 7),
                                                     cosb, sinb, qbb, kbb, vtb, 1280, 16); }
    k_attn<<<dim3(10, 16, 2), 256, 0, stream>>>(qbb, kbb, vtb, catb, 5120, 1280, 16);
    gemm(catb, 0, 5120, 2560, F(ps + 4), 1024, F(ps + 5), tmpo, 0, 1024, 2560, 1024, 5120, 0);
    resid(xbuf, tmpo, mods, 3072, 2048, 1280, 2560);
  }

  // ---- final ----
  dsmall(vecb, 1, F(111), F(112), adab, 0, 0, 2, 2048, 1024);
  ln(xbuf + (size_t)256 * 1024, 1280, 1024, mbuf, F(107), F(108), adab, 2048, 0, 1024);
  gemm(mbuf, 0, 1024, 2048, F(109), 64, F(110), (float*)d_out, 0, 64, 2048, 64, 1024, 0);

  (void)in_sizes; (void)n_in; (void)out_size; (void)ws_size;
}
